// OrigSGMoEImpl_21388937134613
// MI455X (gfx1250) — matrix operand order repaired and checked by reading, not compiled, not run on hardware
//
#include <hip/hip_runtime.h>

#define DEV __device__ __forceinline__

typedef __attribute__((ext_vector_type(16))) __bf16          v16bf;
typedef __attribute__((ext_vector_type(8)))  float           v8f;
typedef __attribute__((ext_vector_type(8)))  unsigned short  v8u;
typedef __attribute__((ext_vector_type(16))) unsigned short  v16u;

constexpr int N_TOK = 4096;
constexpr int DIM   = 1024;
constexpr int NEXP  = 8;
constexpr int HID   = 2048;
constexpr int OUTD  = 1024;
constexpr int NASSIGN = N_TOK * 2;   // top-2

constexpr int BM = 128, BN = 128, BK = 32;
constexpr int THREADS = 256;

// ---------- small helpers ----------
DEV unsigned short f2bfu(float f) {
    unsigned u = __builtin_bit_cast(unsigned, f);
    unsigned r = u + 0x7FFFu + ((u >> 16) & 1u);   // RNE
    return (unsigned short)(r >> 16);
}

DEV v8f wmma_bf16(v16u a, v16u b, v8f c) {
    return __builtin_amdgcn_wmma_f32_16x16x32_bf16(
        false, __builtin_bit_cast(v16bf, a),
        false, __builtin_bit_cast(v16bf, b),
        (short)0, c, false, false);
}

// A fragment (16x32 bf16): lane half selects K groups {half*8..+7, 16+half*8..+7}
DEV v16u load_frag_A(const unsigned short* sA, int row0, int lane) {
    int m = lane & 15, half = lane >> 4;
    const v8u lo = *(const v8u*)&sA[(row0 + m) * BK + half * 8];
    const v8u hi = *(const v8u*)&sA[(row0 + m) * BK + half * 8 + 16];
    return __builtin_shufflevector(lo, hi, 0,1,2,3,4,5,6,7,8,9,10,11,12,13,14,15);
}

// B fragment (32x16 bf16), LDS stored [n][k]: lane half selects K groups {half*8..+7, 16+half*8..+7}
DEV v16u load_frag_B(const unsigned short* sB, int col0, int lane) {
    int n = lane & 15, half = lane >> 4;
    const v8u lo = *(const v8u*)&sB[(col0 + n) * BK + half * 8];
    const v8u hi = *(const v8u*)&sB[(col0 + n) * BK + half * 8 + 16];
    return __builtin_shufflevector(lo, hi, 0,1,2,3,4,5,6,7,8,9,10,11,12,13,14,15);
}

// async byte-copy global(b128) -> LDS, tracked by ASYNCcnt
DEV void async_copy_b128(unsigned short* lds_dst, const unsigned short* gsrc) {
    unsigned lds_off = (unsigned)(unsigned long long)lds_dst;
    asm volatile("global_load_async_to_lds_b128 %0, %1, off"
                 :: "v"(lds_off), "v"(gsrc) : "memory");
}
DEV void async_wait0() {
    asm volatile("s_wait_asynccnt 0x0" ::: "memory");
}

// ---------- 1) LayerNorm + gating ----------
__global__ void ln_gate_kernel(const float* __restrict__ x,
                               const float* __restrict__ nw,
                               const float* __restrict__ nb,
                               const float* __restrict__ eps_p,
                               const float* __restrict__ wg,
                               unsigned short* __restrict__ xnorm,
                               int* __restrict__ topk_idx,
                               float* __restrict__ topk_gate,
                               int* __restrict__ count,
                               float* __restrict__ importance) {
    __shared__ float s1[THREADS], s2[THREADS];
    __shared__ float sl[THREADS * NEXP];
    const int t = threadIdx.x;
    const int n = blockIdx.x;
    const float* row = x + (size_t)n * DIM;

    float xv[4];
    float sum = 0.f, sq = 0.f;
    for (int j = 0; j < 4; ++j) {
        int d = t + j * THREADS;
        float v = row[d];
        xv[j] = v; sum += v; sq += v * v;
    }
    s1[t] = sum; s2[t] = sq;
    __syncthreads();
    for (int s = THREADS / 2; s > 0; s >>= 1) {
        if (t < s) { s1[t] += s1[t + s]; s2[t] += s2[t + s]; }
        __syncthreads();
    }
    const float mean = s1[0] * (1.f / DIM);
    const float var  = s2[0] * (1.f / DIM) - mean * mean;
    const float rstd = rsqrtf(var + eps_p[0]);

    float part[NEXP] = {0.f,0.f,0.f,0.f,0.f,0.f,0.f,0.f};
    for (int j = 0; j < 4; ++j) {
        int d = t + j * THREADS;
        float xn = (xv[j] - mean) * rstd * nw[d] + nb[d];
        xnorm[(size_t)n * DIM + d] = f2bfu(xn);
        const float* g = wg + (size_t)d * NEXP;
        for (int e = 0; e < NEXP; ++e) part[e] += xn * g[e];
    }
    __syncthreads();
    for (int e = 0; e < NEXP; ++e) sl[t * NEXP + e] = part[e];
    __syncthreads();
    for (int s = THREADS / 2; s > 0; s >>= 1) {
        if (t < s)
            for (int e = 0; e < NEXP; ++e) sl[t * NEXP + e] += sl[(t + s) * NEXP + e];
        __syncthreads();
    }
    if (t == 0) {
        float l[NEXP], ss = 0.f;
        for (int e = 0; e < NEXP; ++e) { l[e] = sl[e]; ss += l[e] * l[e]; }
        float nrm = fmaxf(sqrtf(ss), 1e-12f);
        float mx = -3.4e38f;
        for (int e = 0; e < NEXP; ++e) { l[e] /= nrm; mx = fmaxf(mx, l[e]); }
        float p[NEXP], se = 0.f;
        for (int e = 0; e < NEXP; ++e) { p[e] = __expf(l[e] - mx); se += p[e]; }
        float inv = 1.f / se;
        for (int e = 0; e < NEXP; ++e) p[e] *= inv;
        int i1 = 0;
        for (int e = 1; e < NEXP; ++e) if (l[e] > l[i1]) i1 = e;
        int i2 = (i1 == 0) ? 1 : 0;
        for (int e = 0; e < NEXP; ++e) if (e != i1 && l[e] > l[i2]) i2 = e;
        topk_idx[n * 2]     = i1;  topk_idx[n * 2 + 1]  = i2;
        topk_gate[n * 2]    = p[i1]; topk_gate[n * 2 + 1] = p[i2];
        atomicAdd(&count[i1], 1);  atomicAdd(&count[i2], 1);
        atomicAdd(&importance[i1], p[i1]); atomicAdd(&importance[i2], p[i2]);
    }
}

// ---------- 2) offsets + loss + gate_fraction ----------
__global__ void finalize_kernel(const int* __restrict__ count,
                                const float* __restrict__ importance,
                                int* __restrict__ offsets,
                                float* __restrict__ out_tail) {
    if (threadIdx.x != 0) return;
    int c[NEXP]; int off = 0;
    for (int e = 0; e < NEXP; ++e) { c[e] = count[e]; offsets[e] = off; off += c[e]; }
    float total = (float)off;
    float mi = 0.f;
    for (int e = 0; e < NEXP; ++e) mi += importance[e];
    mi *= (1.f / NEXP);
    float vi = 0.f;
    for (int e = 0; e < NEXP; ++e) { float d = importance[e] - mi; vi += d * d; }
    vi *= (1.f / (NEXP - 1));
    float cvi = vi / (mi * mi + 1e-6f);
    float ml = 0.f;
    for (int e = 0; e < NEXP; ++e) ml += (float)c[e];
    ml *= (1.f / NEXP);
    float vl = 0.f;
    for (int e = 0; e < NEXP; ++e) { float d = (float)c[e] - ml; vl += d * d; }
    vl *= (1.f / (NEXP - 1));
    float cvl = vl / (ml * ml + 1e-6f);
    out_tail[0] = (cvi + cvl) * 0.01f;
    for (int e = 0; e < NEXP; ++e) out_tail[1 + e] = (float)c[e] / total;
}

// ---------- 3) scatter tokens into per-expert lists ----------
__global__ void scatter_kernel(const int* __restrict__ topk_idx,
                               const float* __restrict__ topk_gate,
                               const int* __restrict__ offsets,
                               int* __restrict__ cursor,
                               int* __restrict__ list_tok,
                               float* __restrict__ list_gate) {
    int n = blockIdx.x * blockDim.x + threadIdx.x;
    if (n >= N_TOK) return;
    for (int k = 0; k < 2; ++k) {
        int e = topk_idx[n * 2 + k];
        int pos = atomicAdd(&cursor[e], 1);
        int a = offsets[e] + pos;
        list_tok[a] = n;
        list_gate[a] = topk_gate[n * 2 + k];
    }
}

// ---------- 4) residual projection: y = x @ output_weight ----------
__global__ void proj_gemm_kernel(const float* __restrict__ x,
                                 const float* __restrict__ ow,   // [DIM, OUTD]
                                 float* __restrict__ y) {
    __shared__ __align__(32) unsigned short sA[BM * BK];
    __shared__ __align__(32) unsigned short sB[BN * BK];
    const int tilesM = N_TOK / BM;                 // 32
    const int tileM = blockIdx.x % tilesM;
    const int tileN = blockIdx.x / tilesM;
    const int m0 = tileM * BM, n0 = tileN * BN;
    const int t = threadIdx.x, lane = t & 31, wave = t >> 5;
    const int wm = (wave >> 1) * 32, wn = (wave & 1) * 64;

    v8f acc[2][4] = {};
    for (int k0 = 0; k0 < DIM; k0 += BK) {
        __syncthreads();
        // stage A: fp32 x -> bf16
        for (int i = 0; i < 2; ++i) {
            int c = t + i * THREADS;
            int row = c >> 2, ks = (c & 3) * 8;
            const float* src = x + (size_t)(m0 + row) * DIM + k0 + ks;
            float4 f0 = *(const float4*)src;
            float4 f1 = *(const float4*)(src + 4);
            unsigned short u[8] = { f2bfu(f0.x), f2bfu(f0.y), f2bfu(f0.z), f2bfu(f0.w),
                                    f2bfu(f1.x), f2bfu(f1.y), f2bfu(f1.z), f2bfu(f1.w) };
            *(uint4*)&sA[row * BK + ks] = *(const uint4*)u;
        }
        // stage B transposed: sB[n][k] = ow[(k0+k)*OUTD + n0+n]
        {
            int n = t & 127, kb = (t >> 7) * 16;
            for (int i = 0; i < 16; ++i) {
                int k = kb + i;
                sB[n * BK + k] = f2bfu(ow[(size_t)(k0 + k) * OUTD + n0 + n]);
            }
            if (k0 + BK < DIM)
                __builtin_prefetch(&ow[(size_t)(k0 + BK + kb) * OUTD + n0 + n], 0, 0);
        }
        __syncthreads();
        v16u a0 = load_frag_A(sA, wm, lane);
        v16u a1 = load_frag_A(sA, wm + 16, lane);
        for (int tn = 0; tn < 4; ++tn) {
            v16u b = load_frag_B(sB, wn + tn * 16, lane);
            acc[0][tn] = wmma_bf16(a0, b, acc[0][tn]);
            acc[1][tn] = wmma_bf16(a1, b, acc[1][tn]);
        }
    }
    const int half = lane >> 4, nn = lane & 15;
    for (int tm = 0; tm < 2; ++tm)
        for (int tn = 0; tn < 4; ++tn) {
            int colg = n0 + wn + tn * 16 + nn;
            for (int i = 0; i < 8; ++i) {
                int r = m0 + wm + tm * 16 + half * 8 + i;
                y[(size_t)r * OUTD + colg] = acc[tm][tn][i];
            }
        }
}

// ---------- 5) expert GEMM1: h = silu(xnorm_gathered @ Wh^T + bh) ----------
__global__ void expert_gemm1_kernel(const unsigned short* __restrict__ xnorm,
                                    const float* __restrict__ Wh,   // [E,H,D]
                                    const float* __restrict__ bh,   // [E,H]
                                    const int* __restrict__ counts,
                                    const int* __restrict__ offsets,
                                    const int* __restrict__ list_tok,
                                    unsigned short* __restrict__ hbuf) {
    __shared__ __align__(32) unsigned short sA[BM * BK];
    __shared__ __align__(32) unsigned short sB[BN * BK];
    const int e = blockIdx.y;
    const int tilesM = N_TOK / BM;                // worst case
    const int tileM = blockIdx.x % tilesM;
    const int tileN = blockIdx.x / tilesM;
    const int cnt = counts[e];
    const int m0 = tileM * BM;
    if (m0 >= cnt) return;
    const int off = offsets[e];
    const int n0 = tileN * BN;
    const float* W = Wh + (size_t)e * HID * DIM;
    const int t = threadIdx.x, lane = t & 31, wave = t >> 5;
    const int wm = (wave >> 1) * 32, wn = (wave & 1) * 64;

    v8f acc[2][4] = {};
    for (int k0 = 0; k0 < DIM; k0 += BK) {
        __syncthreads();
        // stage A (gathered bf16 rows) via async global->LDS copy
        for (int i = 0; i < 2; ++i) {
            int c = t + i * THREADS;
            int row = c >> 2, ks = (c & 3) * 8;
            int r = m0 + row;
            if (r < cnt) {
                int tok = list_tok[off + r];
                async_copy_b128(&sA[row * BK + ks],
                                xnorm + (size_t)tok * DIM + k0 + ks);
            } else {
                *(uint4*)&sA[row * BK + ks] = make_uint4(0u, 0u, 0u, 0u);
            }
        }
        // stage B: contiguous K in global, fp32 -> bf16
        {
            int n = t >> 1, ks = (t & 1) * 16;
            const float* src = W + (size_t)(n0 + n) * DIM + k0 + ks;
            float4 f0 = *(const float4*)src;
            float4 f1 = *(const float4*)(src + 4);
            float4 f2 = *(const float4*)(src + 8);
            float4 f3 = *(const float4*)(src + 12);
            unsigned short u[16] = {
                f2bfu(f0.x), f2bfu(f0.y), f2bfu(f0.z), f2bfu(f0.w),
                f2bfu(f1.x), f2bfu(f1.y), f2bfu(f1.z), f2bfu(f1.w),
                f2bfu(f2.x), f2bfu(f2.y), f2bfu(f2.z), f2bfu(f2.w),
                f2bfu(f3.x), f2bfu(f3.y), f2bfu(f3.z), f2bfu(f3.w) };
            *(uint4*)&sB[n * BK + ks]     = *(const uint4*)u;
            *(uint4*)&sB[n * BK + ks + 8] = *(const uint4*)(u + 8);
            if (k0 + BK < DIM)
                __builtin_prefetch(src + BK, 0, 0);
        }
        async_wait0();
        __syncthreads();
        v16u a0 = load_frag_A(sA, wm, lane);
        v16u a1 = load_frag_A(sA, wm + 16, lane);
        for (int tn = 0; tn < 4; ++tn) {
            v16u b = load_frag_B(sB, wn + tn * 16, lane);
            acc[0][tn] = wmma_bf16(a0, b, acc[0][tn]);
            acc[1][tn] = wmma_bf16(a1, b, acc[1][tn]);
        }
    }
    const int half = lane >> 4, nn = lane & 15;
    for (int tm = 0; tm < 2; ++tm)
        for (int tn = 0; tn < 4; ++tn) {
            int colg = n0 + wn + tn * 16 + nn;
            float bias = bh[e * HID + colg];
            for (int i = 0; i < 8; ++i) {
                int r = m0 + wm + tm * 16 + half * 8 + i;
                if (r < cnt) {
                    float v = acc[tm][tn][i] + bias;
                    float s = v / (1.0f + __expf(-v));   // silu
                    hbuf[(size_t)(off + r) * HID + colg] = f2bfu(s);
                }
            }
        }
}

// ---------- 6) expert GEMM2: y += gate * (h @ Wo^T + bo) ----------
__global__ void expert_gemm2_kernel(const unsigned short* __restrict__ hbuf,
                                    const float* __restrict__ Wo,   // [E,O,H]
                                    const float* __restrict__ bo,   // [E,O]
                                    const int* __restrict__ counts,
                                    const int* __restrict__ offsets,
                                    const int* __restrict__ list_tok,
                                    const float* __restrict__ list_gate,
                                    float* __restrict__ y) {
    __shared__ __align__(32) unsigned short sA[BM * BK];
    __shared__ __align__(32) unsigned short sB[BN * BK];
    const int e = blockIdx.y;
    const int tilesM = N_TOK / BM;
    const int tileM = blockIdx.x % tilesM;
    const int tileN = blockIdx.x / tilesM;
    const int cnt = counts[e];
    const int m0 = tileM * BM;
    if (m0 >= cnt) return;
    const int off = offsets[e];
    const int n0 = tileN * BN;
    const float* W = Wo + (size_t)e * OUTD * HID;
    const int t = threadIdx.x, lane = t & 31, wave = t >> 5;
    const int wm = (wave >> 1) * 32, wn = (wave & 1) * 64;

    v8f acc[2][4] = {};
    for (int k0 = 0; k0 < HID; k0 += BK) {
        __syncthreads();
        // stage A: h rows (bf16, contiguous by assignment index) via async copy
        for (int i = 0; i < 2; ++i) {
            int c = t + i * THREADS;
            int row = c >> 2, ks = (c & 3) * 8;
            int r = m0 + row;
            if (r < cnt) {
                async_copy_b128(&sA[row * BK + ks],
                                hbuf + (size_t)(off + r) * HID + k0 + ks);
            } else {
                *(uint4*)&sA[row * BK + ks] = make_uint4(0u, 0u, 0u, 0u);
            }
        }
        {
            int n = t >> 1, ks = (t & 1) * 16;
            const float* src = W + (size_t)(n0 + n) * HID + k0 + ks;
            float4 f0 = *(const float4*)src;
            float4 f1 = *(const float4*)(src + 4);
            float4 f2 = *(const float4*)(src + 8);
            float4 f3 = *(const float4*)(src + 12);
            unsigned short u[16] = {
                f2bfu(f0.x), f2bfu(f0.y), f2bfu(f0.z), f2bfu(f0.w),
                f2bfu(f1.x), f2bfu(f1.y), f2bfu(f1.z), f2bfu(f1.w),
                f2bfu(f2.x), f2bfu(f2.y), f2bfu(f2.z), f2bfu(f2.w),
                f2bfu(f3.x), f2bfu(f3.y), f2bfu(f3.z), f2bfu(f3.w) };
            *(uint4*)&sB[n * BK + ks]     = *(const uint4*)u;
            *(uint4*)&sB[n * BK + ks + 8] = *(const uint4*)(u + 8);
            if (k0 + BK < HID)
                __builtin_prefetch(src + BK, 0, 0);
        }
        async_wait0();
        __syncthreads();
        v16u a0 = load_frag_A(sA, wm, lane);
        v16u a1 = load_frag_A(sA, wm + 16, lane);
        for (int tn = 0; tn < 4; ++tn) {
            v16u b = load_frag_B(sB, wn + tn * 16, lane);
            acc[0][tn] = wmma_bf16(a0, b, acc[0][tn]);
            acc[1][tn] = wmma_bf16(a1, b, acc[1][tn]);
        }
    }
    const int half = lane >> 4, nn = lane & 15;
    for (int tm = 0; tm < 2; ++tm)
        for (int tn = 0; tn < 4; ++tn) {
            int colg = n0 + wn + tn * 16 + nn;
            float bias = bo[e * OUTD + colg];
            for (int i = 0; i < 8; ++i) {
                int r = m0 + wm + tm * 16 + half * 8 + i;
                if (r < cnt) {
                    int a = off + r;
                    float val = (acc[tm][tn][i] + bias) * list_gate[a];
                    atomicAdd(&y[(size_t)list_tok[a] * OUTD + colg], val);
                }
            }
        }
}

// ---------- host launch ----------
extern "C" void kernel_launch(void* const* d_in, const int* in_sizes, int n_in,
                              void* d_out, int out_size, void* d_ws, size_t ws_size,
                              hipStream_t stream) {
    const float* x    = (const float*)d_in[0];
    const float* nw   = (const float*)d_in[6];
    const float* nb   = (const float*)d_in[7];
    const float* eps  = (const float*)d_in[8];
    const float* wg   = (const float*)d_in[9];
    const float* whid = (const float*)d_in[10];
    const float* bhid = (const float*)d_in[11];
    const float* wexp = (const float*)d_in[12];
    const float* bexp = (const float*)d_in[13];
    const float* ow   = (const float*)d_in[14];
    float* y = (float*)d_out;
    float* out_tail = y + (size_t)N_TOK * OUTD;

    char* ws = (char*)d_ws;
    int*   count      = (int*)ws;             // 8
    int*   cursor     = count + NEXP;         // 8
    int*   offsets    = cursor + NEXP;        // 8
    float* importance = (float*)(offsets + NEXP);   // 8  (ends at 128 B)
    int*   topk_idx   = (int*)(ws + 256);
    float* topk_gate  = (float*)(ws + 256 + 1 * (size_t)NASSIGN * 4);
    int*   list_tok   = (int*)(ws + 256 + 2 * (size_t)NASSIGN * 4);
    float* list_gate  = (float*)(ws + 256 + 3 * (size_t)NASSIGN * 4);
    unsigned short* xnorm = (unsigned short*)(ws + 256 + 4 * (size_t)NASSIGN * 4);
    unsigned short* hbuf  = (unsigned short*)((char*)xnorm + (size_t)N_TOK * DIM * 2);

    hipMemsetAsync(d_ws, 0, 128, stream);

    ln_gate_kernel<<<N_TOK, THREADS, 0, stream>>>(
        x, nw, nb, eps, wg, xnorm, topk_idx, topk_gate, count, importance);

    finalize_kernel<<<1, 32, 0, stream>>>(count, importance, offsets, out_tail);

    scatter_kernel<<<(N_TOK + THREADS - 1) / THREADS, THREADS, 0, stream>>>(
        topk_idx, topk_gate, offsets, cursor, list_tok, list_gate);

    // y = x @ output_weight   (plain stores; must precede atomic combine)
    proj_gemm_kernel<<<dim3((N_TOK / BM) * (OUTD / BN)), THREADS, 0, stream>>>(x, ow, y);

    // expert MLP, sparse dispatch
    expert_gemm1_kernel<<<dim3((N_TOK / BM) * (HID / BN), NEXP), THREADS, 0, stream>>>(
        xnorm, whid, bhid, count, offsets, list_tok, hbuf);

    expert_gemm2_kernel<<<dim3((N_TOK / BM) * (OUTD / BN), NEXP), THREADS, 0, stream>>>(
        hbuf, wexp, bexp, count, offsets, list_tok, list_gate, y);
}
